// _GCNDrugEncoder_71708773974504
// MI455X (gfx1250) — matrix operand order repaired and checked by reading, not compiled, not run on hardware
//
#include <hip/hip_runtime.h>
#include <hip/hip_bf16.h>

#define N_NODES 100000
#define N_EDGES 300000
#define HIDDEN  256
#define IN_DIM  82
#define K1PAD   96   // IN_DIM padded to multiple of 32 for the WMMA K-loop

typedef __attribute__((ext_vector_type(16))) __bf16         bf16x16;
typedef __attribute__((ext_vector_type(8)))  float          floatx8;
typedef __attribute__((ext_vector_type(8)))  unsigned short u16x8;

union Frag {
    u16x8   h[2];
    bf16x16 v;
};

// fp32 -> bf16 round-to-nearest-even (bit trick)
__device__ __forceinline__ unsigned short f2bf(float f) {
    unsigned int u = __float_as_uint(f);
    u += 0x7FFFu + ((u >> 16) & 1u);
    return (unsigned short)(u >> 16);
}

// 32-bit wave-relative LDS address: low 32 bits of the LDS-aperture flat addr
__device__ __forceinline__ unsigned lds_addr(const void* p) {
    return (unsigned)(unsigned long long)(size_t)p;
}

// CDNA5 async copy: global -> LDS, 16 bytes per lane, tracked by ASYNCcnt
__device__ __forceinline__ void async_b128(const void* g, unsigned ldsoff) {
    asm volatile("global_load_async_to_lds_b128 %0, %1, off"
                 :: "v"(ldsoff), "v"((unsigned long long)(size_t)g)
                 : "memory");
}

__device__ __forceinline__ void wait_async0() {
    asm volatile("s_wait_asynccnt 0x0" ::: "memory");
}

// ---------------------------------------------------------------- utilities
__global__ void k_zero_f32(float* __restrict__ p, int n) {
    int i = blockIdx.x * blockDim.x + threadIdx.x;
    if (i < n) p[i] = 0.0f;
}

__global__ void k_degree(const int* __restrict__ dst, float* __restrict__ deg) {
    int e = blockIdx.x * blockDim.x + threadIdx.x;
    if (e < N_EDGES) atomicAdd(&deg[dst[e]], 1.0f);
}

__global__ void k_dinv(const float* __restrict__ deg, float* __restrict__ dinv) {
    int i = blockIdx.x * blockDim.x + threadIdx.x;
    if (i < N_NODES) dinv[i] = rsqrtf(deg[i] + 1.0f);  // +1 for self loop
}

// activations: [rows, cols] fp32 -> [rows, colsPad] bf16 (zero padded cols)
__global__ void k_cvt_act(const float* __restrict__ src, unsigned short* __restrict__ dst,
                          int rows, int cols, int colsPad) {
    int i = blockIdx.x * blockDim.x + threadIdx.x;
    if (i >= rows * colsPad) return;
    int r = i / colsPad, c = i - r * colsPad;
    dst[i] = (c < cols) ? f2bf(src[r * cols + c]) : (unsigned short)0;
}

// weights: [krows, 256] fp32 -> [kpad, 256] bf16 (zero padded rows)
__global__ void k_cvt_w(const float* __restrict__ src, unsigned short* __restrict__ dst,
                        int krows, int kpad) {
    int i = blockIdx.x * blockDim.x + threadIdx.x;
    if (i >= kpad * HIDDEN) return;
    int r = i / HIDDEN, c = i - r * HIDDEN;
    dst[i] = (r < krows) ? f2bf(src[r * HIDDEN + c]) : (unsigned short)0;
}

// ---------------------------------------------------------------- WMMA GEMM
// C[M,256] = A[M,K](bf16) @ B[K,256](bf16), fp32 accumulate.
// Block: 256 threads = 8 waves. Block tile: 32 rows x 128 cols.
// Double-buffered LDS, tiles staged with GLOBAL_LOAD_ASYNC_TO_LDS_B128.
// K is a compile-time constant (96 or 256) so the loop fully unrolls and
// LDS/global offsets fold into instruction immediates.
// grid = (M/32, 256/128).
#define BUF_EL 5120   // elements per buffer: A 32x32 (1024) + B 32x128 (4096)

template<int K>
__global__ __launch_bounds__(256) void k_gemm_wmma(
    const unsigned short* __restrict__ A, const unsigned short* __restrict__ B,
    float* __restrict__ C) {
    __shared__ unsigned short smem[2 * BUF_EL];   // 20 KB
    const int t    = threadIdx.x;
    const int lane = t & 31;
    const int wave = t >> 5;
    const int rowBase = blockIdx.x * 32;
    const int colBase = blockIdx.y * 128;
    const int r  = lane & 15;   // fragment row / col within tile
    const int hh = lane >> 4;   // K-half selector

    // staging geometry (16B chunks)
    const int br0  = t >> 4;          // B rows 0..15 (chunk 0); +16 for chunk 1
    const int boff = (t & 15) * 8;    // element offset within the 128-wide B row
    const int arow = t >> 2;          // A row 0..31 (threads 0..127 only)
    const int aoff = (t & 3) * 8;     // A element offset

    // loop-invariant staging base pointers / LDS offsets
    const unsigned short* bg = B + (size_t)br0 * HIDDEN + colBase + boff;
    const unsigned short* ag = A + (size_t)(rowBase + arow) * K + aoff;
    const unsigned bl = lds_addr(&smem[1024 + br0 * 128 + boff]);
    const unsigned al = lds_addr(&smem[arow * 32 + aoff]);

    floatx8 acc0 = {};
    floatx8 acc1 = {};

    auto stage = [&](int kk, int bufByte) {
        const unsigned short* b0 = bg + (size_t)kk * HIDDEN;
        async_b128(b0, bl + bufByte);
        async_b128(b0 + (size_t)16 * HIDDEN, bl + bufByte + 16 * 128 * 2);
        if (t < 128) async_b128(ag + kk, al + bufByte);
    };

    stage(0, 0);
    #pragma unroll
    for (int kk = 0; kk < K; kk += 32) {
        const int cur = (kk >> 5) & 1;             // compile-time after unroll
        const int bufEl = cur * BUF_EL;
        wait_async0();        // this wave's staged tile has landed in LDS
        __syncthreads();      // everyone's tile has landed
        if (kk + 32 < K) stage(kk + 32, (BUF_EL - bufEl) * 2);  // prefetch next

        const unsigned short* As = smem + bufEl;
        const unsigned short* Bs = smem + bufEl + 1024;

        // B fragment: lane holds col n = wave*16 + r, k = hh*8+e (e<8) and 16+hh*8+e
        Frag bf;
        {
            const unsigned short* p = Bs + (hh * 8) * 128 + wave * 16 + r;
            #pragma unroll
            for (int e = 0; e < 8; ++e) {
                bf.h[0][e] = p[e * 128];
                bf.h[1][e] = p[(16 + e) * 128];
            }
        }
        // A fragments: lane holds row r, k = hh*8+e (e<8) and 16+hh*8+e
        Frag a0, a1;
        {
            const u16x8* p0 = (const u16x8*)(As + r * 32 + hh * 8);
            a0.h[0] = p0[0];
            a0.h[1] = p0[2];   // +16 elements
            const u16x8* p1 = (const u16x8*)(As + (16 + r) * 32 + hh * 8);
            a1.h[0] = p1[0];
            a1.h[1] = p1[2];
        }
        acc0 = __builtin_amdgcn_wmma_f32_16x16x32_bf16(false, a0.v, false, bf.v,
                                                       (short)0, acc0, false, false);
        acc1 = __builtin_amdgcn_wmma_f32_16x16x32_bf16(false, a1.v, false, bf.v,
                                                       (short)0, acc1, false, false);
        __syncthreads();      // all reads of this buffer done before reuse
    }

    // store C: VGPR v, lane l -> row = v + 8*(l>>4), col = l&15
    const int col  = colBase + wave * 16 + r;
    const int rsel = 8 * hh;
    #pragma unroll
    for (int v = 0; v < 8; ++v) {
        C[(size_t)(rowBase + v + rsel) * HIDDEN + col]      = acc0[v];
        C[(size_t)(rowBase + 16 + v + rsel) * HIDDEN + col] = acc1[v];
    }
}

// ---------------------------------------------------------------- edge scatter
// agg[dst] += xW[src] * dinv[src]*dinv[dst]; one wave per edge, 8 cols per lane
__global__ __launch_bounds__(256) void k_aggregate(
    const int* __restrict__ src, const int* __restrict__ dst,
    const float* __restrict__ dinv, const float* __restrict__ xW,
    float* __restrict__ agg) {
    int wave = threadIdx.x >> 5, lane = threadIdx.x & 31;
    int e = blockIdx.x * 8 + wave;
    if (e >= N_EDGES) return;
    int s = src[e], d = dst[e];
    float norm = dinv[s] * dinv[d];
    const float4* xs = (const float4*)(xW + (size_t)s * HIDDEN) + lane * 2;
    float* ad = agg + (size_t)d * HIDDEN + lane * 8;
    float4 v0 = xs[0], v1 = xs[1];
    atomicAdd(ad + 0, v0.x * norm);
    atomicAdd(ad + 1, v0.y * norm);
    atomicAdd(ad + 2, v0.z * norm);
    atomicAdd(ad + 3, v0.w * norm);
    atomicAdd(ad + 4, v1.x * norm);
    atomicAdd(ad + 5, v1.y * norm);
    atomicAdd(ad + 6, v1.z * norm);
    atomicAdd(ad + 7, v1.w * norm);
}

// h = relu(agg + xW*dinv^2 + b); writes bf16 (next layer input) and optional fp32
__global__ __launch_bounds__(256) void k_act(
    const float* __restrict__ agg, const float* __restrict__ xW,
    const float* __restrict__ dinv, const float* __restrict__ b,
    unsigned short* __restrict__ hbf, float* __restrict__ hf32, int writeF32) {
    int n = blockIdx.x, c = threadIdx.x;
    size_t i = (size_t)n * HIDDEN + c;
    float di = dinv[n];
    float v = agg[i] + xW[i] * di * di + b[c];
    v = fmaxf(v, 0.0f);
    hbf[i] = f2bf(v);
    if (writeF32) hf32[i] = v;
}

// out[batch[n]] = max(out[batch[n]], h[n]); h >= 0 so uint bit-pattern max is exact
__global__ __launch_bounds__(256) void k_segmax(
    const float* __restrict__ h, const int* __restrict__ batch,
    float* __restrict__ out) {
    int n = blockIdx.x, c = threadIdx.x;
    int g = batch[n];
    unsigned int val = __float_as_uint(h[(size_t)n * HIDDEN + c]);
    atomicMax((unsigned int*)(out + (size_t)g * HIDDEN + c), val);
}

// ---------------------------------------------------------------- launch
extern "C" void kernel_launch(void* const* d_in, const int* in_sizes, int n_in,
                              void* d_out, int out_size, void* d_ws, size_t ws_size,
                              hipStream_t stream) {
    const float* x     = (const float*)d_in[0];
    const int*   edge  = (const int*)d_in[1];   // [2, E]
    const int*   batch = (const int*)d_in[2];
    const float* W1    = (const float*)d_in[4];
    const float* b1    = (const float*)d_in[5];
    const float* W2    = (const float*)d_in[6];
    const float* b2    = (const float*)d_in[7];
    const float* W3    = (const float*)d_in[8];
    const float* b3    = (const float*)d_in[9];
    const int* esrc = edge;
    const int* edst = edge + N_EDGES;

    // carve workspace (256B aligned slabs)
    char* p = (char*)d_ws;
    auto carve = [&](size_t bytes) {
        char* r = p;
        p += (bytes + 255) & ~(size_t)255;
        return r;
    };
    float*          deg  = (float*)carve((size_t)N_NODES * 4);
    float*          dinv = (float*)carve((size_t)N_NODES * 4);
    unsigned short* xpad = (unsigned short*)carve((size_t)N_NODES * K1PAD * 2);
    unsigned short* w1b  = (unsigned short*)carve((size_t)K1PAD * HIDDEN * 2);
    unsigned short* w2b  = (unsigned short*)carve((size_t)HIDDEN * HIDDEN * 2);
    unsigned short* w3b  = (unsigned short*)carve((size_t)HIDDEN * HIDDEN * 2);
    float*          xW   = (float*)carve((size_t)N_NODES * HIDDEN * 4);
    float*          agg  = (float*)carve((size_t)N_NODES * HIDDEN * 4);
    unsigned short* hbf  = (unsigned short*)carve((size_t)N_NODES * HIDDEN * 2);
    (void)ws_size; (void)in_sizes; (void)n_in;

    const int TB = 256;
    // degrees + normalization
    k_zero_f32<<<(N_NODES + TB - 1) / TB, TB, 0, stream>>>(deg, N_NODES);
    k_degree  <<<(N_EDGES + TB - 1) / TB, TB, 0, stream>>>(edst, deg);
    k_dinv    <<<(N_NODES + TB - 1) / TB, TB, 0, stream>>>(deg, dinv);

    // precision conversion (fp32 -> bf16, K padded to 96 for layer 1)
    k_cvt_act<<<((size_t)N_NODES * K1PAD + TB - 1) / TB, TB, 0, stream>>>(x, xpad, N_NODES, IN_DIM, K1PAD);
    k_cvt_w  <<<(K1PAD * HIDDEN + TB - 1) / TB, TB, 0, stream>>>(W1, w1b, IN_DIM, K1PAD);
    k_cvt_w  <<<(HIDDEN * HIDDEN + TB - 1) / TB, TB, 0, stream>>>(W2, w2b, HIDDEN, HIDDEN);
    k_cvt_w  <<<(HIDDEN * HIDDEN + TB - 1) / TB, TB, 0, stream>>>(W3, w3b, HIDDEN, HIDDEN);

    dim3 ggrid(N_NODES / 32, HIDDEN / 128);             // 3125 x 2
    int  zgrid = (N_NODES * HIDDEN + TB - 1) / TB;      // zero agg
    int  agrid = N_EDGES / 8;                           // 8 edges per block

    // ---- layer 1
    k_gemm_wmma<K1PAD><<<ggrid, TB, 0, stream>>>(xpad, w1b, xW);
    k_zero_f32 <<<zgrid, TB, 0, stream>>>(agg, N_NODES * HIDDEN);
    k_aggregate<<<agrid, TB, 0, stream>>>(esrc, edst, dinv, xW, agg);
    k_act      <<<N_NODES, TB, 0, stream>>>(agg, xW, dinv, b1, hbf, nullptr, 0);

    // ---- layer 2
    k_gemm_wmma<HIDDEN><<<ggrid, TB, 0, stream>>>(hbf, w2b, xW);
    k_zero_f32 <<<zgrid, TB, 0, stream>>>(agg, N_NODES * HIDDEN);
    k_aggregate<<<agrid, TB, 0, stream>>>(esrc, edst, dinv, xW, agg);
    k_act      <<<N_NODES, TB, 0, stream>>>(agg, xW, dinv, b2, hbf, nullptr, 0);

    // ---- layer 3 (fp32 activations written in-place into agg)
    k_gemm_wmma<HIDDEN><<<ggrid, TB, 0, stream>>>(hbf, w3b, xW);
    k_zero_f32 <<<zgrid, TB, 0, stream>>>(agg, N_NODES * HIDDEN);
    k_aggregate<<<agrid, TB, 0, stream>>>(esrc, edst, dinv, xW, agg);
    k_act      <<<N_NODES, TB, 0, stream>>>(agg, xW, dinv, b3, hbf, agg, 1);

    // ---- graph-wise max pooling
    k_zero_f32<<<(out_size + TB - 1) / TB, TB, 0, stream>>>((float*)d_out, out_size);
    k_segmax  <<<N_NODES, TB, 0, stream>>>(agg, batch, (float*)d_out);
}
